// DoubleRNNAE_24215025615652
// MI455X (gfx1250) — compile-verified, matrix operand order checked by reading, not run on hardware
//
#include <hip/hip_runtime.h>
#include <hip/hip_bf16.h>

// ---------------------------------------------------------------------------
// Persistent LSTM seq2seq autoencoder for MI455X (gfx1250, wave32, WMMA).
//   B=64, T=2048, D=128, H=256.  16 persistent workgroups, 512 thr (16 waves).
//   Each WG owns a 16-wide hidden slice -> 64 gate rows (i,f,g,o x 16).
//   Weights converted fp32->bf16 once per phase into LDS (WMMA B-fragment
//   layout). Per step: stage A=[x_t | h] as bf16 A-fragments in LDS, each
//   wave runs 12x v_wmma_f32_16x16x32_bf16, gate nonlinearity, h ping-pong
//   through global, device-scope software barrier.
// ---------------------------------------------------------------------------

#define BATCH   64
#define SEQT    2048
#define THALF   1024
#define DIN     128
#define HID     256

#define NWG      16
#define NTHREADS 512
#define JSLICE   16          // HID / NWG

typedef __bf16 bf16_t;
typedef __bf16 v16bf __attribute__((ext_vector_type(16)));
typedef float  v8f   __attribute__((ext_vector_type(8)));

#define SA_ELEMS  (4 * 12 * 512)     // 48 A-tiles (4 mt x 12 kt), 512 bf16 each
#define SW_ELEMS  (4 * 12 * 512)     // 48 B-tiles (4 gate-ntiles x 12 kt)
#define SWL_ELEMS (8 * 512)          // projection B-tiles (8 kt)
// byte offsets inside dynamic LDS
#define OFF_SA    0
#define OFF_SW    (OFF_SA  + SA_ELEMS * 2)      // 49152
#define OFF_SWL   (OFF_SW  + SW_ELEMS * 2)      // 98304
#define OFF_GATE  (OFF_SWL + SWL_ELEMS * 2)     // 106496
#define OFF_C     (OFF_GATE + 4096 * 4)         // 122880
#define OFF_RED   (OFF_C    + 1024 * 4)         // 126976
#define OFF_BIAS  (OFF_RED  + 1024 * 4)         // 131072
#define OFF_BL    (OFF_BIAS + 64 * 4)           // 131328
#define SMEM_BYTES (OFF_BL + 16 * 4)            // 131392

// fragment element (lane-hi, e) -> K index inside a 32-wide K tile
// (matches ISA 16-bit A/B layout: VGPR0 lanes0-15 K=0,1; lanes16-31 K=8,9; ...)
__device__ __forceinline__ int kfrag(int hi, int e) {
    return ((e & 7) | ((e >> 3) << 4)) + (hi << 3);
}

__device__ __forceinline__ float sigm(float v) { return 1.f / (1.f + __expf(-v)); }

__global__ void init_ws_kernel(float* hbuf, unsigned* bar) {
    int i = blockIdx.x * blockDim.x + threadIdx.x;
    for (int q = i; q < 2 * BATCH * HID; q += gridDim.x * blockDim.x) hbuf[q] = 0.f;
    if (i == 0) { bar[0] = 0u; bar[1] = 0u; }
}

__global__ __launch_bounds__(NTHREADS)
void rnn_ae_persistent(
    const float* __restrict__ x,
    const float* e1_Wih, const float* e1_Whh, const float* e1_bih, const float* e1_bhh,
    const float* e2_Wih, const float* e2_Whh, const float* e2_bih, const float* e2_bhh,
    const float* d1_Wih, const float* d1_Whh, const float* d1_bih, const float* d1_bhh,
    const float* d2_Wih, const float* d2_Whh, const float* d2_bih, const float* d2_bhh,
    const float* l1_W, const float* l1_b, const float* l2_W, const float* l2_b,
    float* __restrict__ out,
    float* hbuf, float* inpbuf, float* h1, float* c1, float* h2, float* c2,
    unsigned* bar)
{
    extern __shared__ char smem[];
    bf16_t* sA    = (bf16_t*)(smem + OFF_SA);
    bf16_t* sW    = (bf16_t*)(smem + OFF_SW);
    bf16_t* sWl   = (bf16_t*)(smem + OFF_SWL);
    float*  sGate = (float*)(smem + OFF_GATE);
    float*  sC    = (float*)(smem + OFF_C);
    float*  sRed  = (float*)(smem + OFF_RED);
    float*  sBias = (float*)(smem + OFF_BIAS);
    float*  sBl   = (float*)(smem + OFF_BL);

    const int tid  = threadIdx.x;
    const int wv   = tid >> 5;
    const int lane = tid & 31;
    const int wg   = blockIdx.x;
    const int j0   = wg * JSLICE;      // hidden slice owned by this WG
    int hp = 0;                        // current h lives in hbuf slot hp

    // ---- device-scope software barrier over the NWG persistent workgroups
    auto gbar = [&]() {
        __syncthreads();
        if (tid == 0) {
            __threadfence();
            unsigned g = __hip_atomic_load(&bar[1], __ATOMIC_RELAXED, __HIP_MEMORY_SCOPE_AGENT);
            unsigned p = __hip_atomic_fetch_add(&bar[0], 1u, __ATOMIC_ACQ_REL, __HIP_MEMORY_SCOPE_AGENT);
            if (p == (unsigned)(NWG - 1)) {
                __hip_atomic_store(&bar[0], 0u, __ATOMIC_RELAXED, __HIP_MEMORY_SCOPE_AGENT);
                __hip_atomic_fetch_add(&bar[1], 1u, __ATOMIC_RELEASE, __HIP_MEMORY_SCOPE_AGENT);
            } else {
                while (__hip_atomic_load(&bar[1], __ATOMIC_ACQUIRE, __HIP_MEMORY_SCOPE_AGENT) == g)
                    __builtin_amdgcn_s_sleep(2);
            }
            __threadfence();
        }
        __syncthreads();
    };

    // ---- pack this WG's 64 gate rows of [Wih | Whh] as bf16 B-fragments
    auto load_gate_weights = [&](const float* Wih, const float* Whh,
                                 const float* bih, const float* bhh) {
        for (int idx = tid; idx < SW_ELEMS; idx += NTHREADS) {
            int g   = idx / 6144;
            int rem = idx - g * 6144;
            int kt  = rem >> 9;
            int ln  = (rem >> 4) & 31;
            int e   = rem & 15;
            int n   = ln & 15, hi = ln >> 4;
            int k   = kt * 32 + kfrag(hi, e);
            int r   = g * HID + j0 + n;
            float v = (k < DIN) ? Wih[(long)r * DIN + k] : Whh[(long)r * HID + (k - DIN)];
            sW[idx] = (bf16_t)v;
        }
        for (int idx = tid; idx < 64; idx += NTHREADS) {
            int g = idx >> 4, n = idx & 15;
            int r = g * HID + j0 + n;
            sBias[idx] = bih[r] + bhh[r];
        }
    };

    // ---- stage A = [x_t | h] as bf16 A-fragments (4 mt x 12 kt tiles)
    auto fill_sA_full = [&](const float* xsrc, long xstride) {
        const float* hc = hbuf + (long)hp * (BATCH * HID);
        for (int idx = tid; idx < SA_ELEMS; idx += NTHREADS) {
            int mt  = idx / 6144;
            int rem = idx - mt * 6144;
            int kt  = rem >> 9;
            int ln  = (rem >> 4) & 31;
            int e   = rem & 15;
            int m   = mt * 16 + (ln & 15);
            int hi  = ln >> 4;
            int k   = kt * 32 + kfrag(hi, e);
            float v = (k < DIN) ? xsrc[(long)m * xstride + k]
                                : hc[(long)m * HID + (k - DIN)];
            sA[idx] = (bf16_t)v;
        }
    };

    // ---- one LSTM cell step: 12 chained bf16 WMMAs per wave + gate math
    auto lstm_step = [&]() {
        int mt = wv & 3, g = wv >> 2;
        v8f acc = {0.f, 0.f, 0.f, 0.f, 0.f, 0.f, 0.f, 0.f};
#pragma unroll
        for (int kt = 0; kt < 12; ++kt) {
            v16bf a = *(const v16bf*)(sA + ((mt * 12 + kt) * 32 + lane) * 16);
            v16bf b = *(const v16bf*)(sW + ((g * 12 + kt) * 32 + lane) * 16);
            acc = __builtin_amdgcn_wmma_f32_16x16x32_bf16(
                false, a, false, b, (short)0, acc, false, false);
        }
        int hi = lane >> 4, n = lane & 15;
#pragma unroll
        for (int r = 0; r < 8; ++r) {
            int m = mt * 16 + hi * 8 + r;
            sGate[g * 1024 + m * 16 + n] = acc[r] + sBias[g * 16 + n];
        }
        __syncthreads();
        float* hn = hbuf + (long)(hp ^ 1) * (BATCH * HID);
        for (int q = tid; q < 1024; q += NTHREADS) {
            int m = q >> 4, n2 = q & 15;
            float gi = sGate[q],        gf = sGate[1024 + q];
            float gg = sGate[2048 + q], go = sGate[3072 + q];
            float c = sigm(gf) * sC[q] + sigm(gi) * tanhf(gg);
            sC[q] = c;
            hn[(long)m * HID + j0 + n2] = sigm(go) * tanhf(c);
        }
    };

    auto save_state = [&](float* hdst, float* cdst) {
        const float* hc = hbuf + (long)hp * (BATCH * HID);
        for (int q = tid; q < 1024; q += NTHREADS) {
            int m = q >> 4, n = q & 15;
            hdst[(long)m * HID + j0 + n] = hc[(long)m * HID + j0 + n];
            cdst[(long)m * HID + j0 + n] = sC[q];
        }
    };
    auto restore_state = [&](const float* hsrc, const float* csrc) {
        float* hc = hbuf + (long)hp * (BATCH * HID);
        for (int q = tid; q < 1024; q += NTHREADS) {
            int m = q >> 4, n = q & 15;
            hc[(long)m * HID + j0 + n] = hsrc[(long)m * HID + j0 + n];
            sC[q] = csrc[(long)m * HID + j0 + n];
        }
    };

    auto encode_phase = [&](const float* Wih, const float* Whh,
                            const float* bih, const float* bhh, int t0, int tstep) {
        load_gate_weights(Wih, Whh, bih, bhh);
        __syncthreads();
        for (int s = 0; s < THALF; ++s) {
            int t = t0 + tstep * s;
            // prefetch next timestep's rows of x into cache
            if (s + 1 < THALF) {
                int tn = t + tstep;
                for (int q = tid; q < BATCH; q += NTHREADS)
                    __builtin_prefetch(x + ((long)q * SEQT + tn) * DIN, 0, 1);
            }
            fill_sA_full(x + (long)t * DIN, (long)SEQT * DIN);
            __syncthreads();
            lstm_step();
            hp ^= 1;
            gbar();
        }
    };

    auto decode_phase = [&](const float* Wih, const float* Whh,
                            const float* bih, const float* bhh,
                            const float* Wl, const float* bl, int t0, int tstep) {
        const int ct = wg & 7;  // 16-col slice of the 128-wide projection output
        load_gate_weights(Wih, Whh, bih, bhh);
        for (int idx = tid; idx < SWL_ELEMS; idx += NTHREADS) {
            int kt = idx >> 9;
            int ln = (idx >> 4) & 31;
            int e  = idx & 15;
            int n  = ln & 15, hi = ln >> 4;
            int k  = kt * 32 + kfrag(hi, e);
            sWl[idx] = (bf16_t)Wl[(long)(ct * 16 + n) * HID + k];
        }
        for (int q = tid; q < 16; q += NTHREADS) sBl[q] = bl[ct * 16 + q];
        __syncthreads();

        for (int s = 0; s < THALF; ++s) {
            // refresh the h-part of the A-fragments from the current h
            const float* hc = hbuf + (long)hp * (BATCH * HID);
            for (int idx = tid; idx < 4 * 8 * 512; idx += NTHREADS) {
                int mt  = idx / 4096;
                int rem = idx - mt * 4096;
                int kt  = 4 + (rem >> 9);
                int ln  = (rem >> 4) & 31;
                int e   = rem & 15;
                int m   = mt * 16 + (ln & 15);
                int hi  = ln >> 4;
                int k   = kt * 32 + kfrag(hi, e);
                sA[((mt * 12 + kt) * 32 + ln) * 16 + e] =
                    (bf16_t)hc[(long)m * HID + (k - DIN)];
            }
            for (int q = tid; q < 1024; q += NTHREADS) sRed[q] = 0.f;
            __syncthreads();

            // projection: inp = h @ Wl^T + bl  (K=256 split over 4 wave groups)
            int mt = wv & 3, ks = wv >> 2;
            v8f acc = {0.f, 0.f, 0.f, 0.f, 0.f, 0.f, 0.f, 0.f};
#pragma unroll
            for (int kk = 0; kk < 2; ++kk) {
                int kp = ks * 2 + kk;
                v16bf a = *(const v16bf*)(sA + ((mt * 12 + 4 + kp) * 32 + lane) * 16);
                v16bf b = *(const v16bf*)(sWl + (kp * 32 + lane) * 16);
                acc = __builtin_amdgcn_wmma_f32_16x16x32_bf16(
                    false, a, false, b, (short)0, acc, false, false);
            }
            int hi = lane >> 4, n = lane & 15;
#pragma unroll
            for (int r = 0; r < 8; ++r)
                atomicAdd(&sRed[mt * 256 + (hi * 8 + r) * 16 + n], acc[r]);
            __syncthreads();

            int t_out = t0 + tstep * s;
            for (int q = tid; q < 1024; q += NTHREADS) {
                int mt2 = q >> 8, w2 = q & 255;
                int m = mt2 * 16 + (w2 >> 4), n2 = w2 & 15;
                float v = sRed[q] + sBl[n2];
                int col = ct * 16 + n2;
                inpbuf[(long)m * DIN + col] = v;
                out[((long)m * SEQT + t_out) * DIN + col] = v;
            }
            gbar();
            if (s == THALF - 1) break;

            fill_sA_full(inpbuf, (long)DIN);
            __syncthreads();
            lstm_step();
            hp ^= 1;
            gbar();
        }
    };

    // ------------------------------ run -----------------------------------
    for (int q = tid; q < 1024; q += NTHREADS) sC[q] = 0.f;  // c0 = 0 (h0 zeroed by init kernel)

    encode_phase(e1_Wih, e1_Whh, e1_bih, e1_bhh, THALF - 1, -1);  // x[:, :T1][::-1]
    save_state(h1, c1);
    gbar();

    encode_phase(e2_Wih, e2_Whh, e2_bih, e2_bhh, THALF, +1);      // x[:, T1:]
    save_state(h2, c2);
    gbar();

    restore_state(h1, c1);
    gbar();
    decode_phase(d1_Wih, d1_Whh, d1_bih, d1_bhh, l1_W, l1_b, 0, +1);

    restore_state(h2, c2);
    gbar();
    decode_phase(d2_Wih, d2_Whh, d2_bih, d2_bhh, l2_W, l2_b, SEQT - 1, -1);
}

extern "C" void kernel_launch(void* const* d_in, const int* in_sizes, int n_in,
                              void* d_out, int out_size, void* d_ws, size_t ws_size,
                              hipStream_t stream) {
    (void)in_sizes; (void)n_in; (void)out_size; (void)ws_size;

    const float* x      = (const float*)d_in[0];
    const float* e1_Wih = (const float*)d_in[1];
    const float* e1_Whh = (const float*)d_in[2];
    const float* e1_bih = (const float*)d_in[3];
    const float* e1_bhh = (const float*)d_in[4];
    const float* e2_Wih = (const float*)d_in[5];
    const float* e2_Whh = (const float*)d_in[6];
    const float* e2_bih = (const float*)d_in[7];
    const float* e2_bhh = (const float*)d_in[8];
    const float* d1_Wih = (const float*)d_in[9];
    const float* d1_Whh = (const float*)d_in[10];
    const float* d1_bih = (const float*)d_in[11];
    const float* d1_bhh = (const float*)d_in[12];
    const float* d2_Wih = (const float*)d_in[13];
    const float* d2_Whh = (const float*)d_in[14];
    const float* d2_bih = (const float*)d_in[15];
    const float* d2_bhh = (const float*)d_in[16];
    const float* l1_W   = (const float*)d_in[17];
    const float* l1_b   = (const float*)d_in[18];
    const float* l2_W   = (const float*)d_in[19];
    const float* l2_b   = (const float*)d_in[20];

    float* out = (float*)d_out;

    // workspace layout (floats)
    float* ws     = (float*)d_ws;
    float* hbuf   = ws;                 // 2 * 64*256  (ping-pong h)
    float* inpbuf = ws + 32768;         // 64*128
    float* h1     = ws + 40960;         // 64*256
    float* c1     = ws + 57344;
    float* h2     = ws + 73728;
    float* c2     = ws + 90112;
    unsigned* bar = (unsigned*)(ws + 106496);

    (void)hipFuncSetAttribute((const void*)rnn_ae_persistent,
                              hipFuncAttributeMaxDynamicSharedMemorySize,
                              (int)SMEM_BYTES);

    hipLaunchKernelGGL(init_ws_kernel, dim3(64), dim3(256), 0, stream, hbuf, bar);
    hipLaunchKernelGGL(rnn_ae_persistent, dim3(NWG), dim3(NTHREADS), SMEM_BYTES, stream,
                       x,
                       e1_Wih, e1_Whh, e1_bih, e1_bhh,
                       e2_Wih, e2_Whh, e2_bih, e2_bhh,
                       d1_Wih, d1_Whh, d1_bih, d1_bhh,
                       d2_Wih, d2_Whh, d2_bih, d2_bhh,
                       l1_W, l1_b, l2_W, l2_b,
                       out, hbuf, inpbuf, h1, c1, h2, c2, bar);
}
